// AcousticModel_45887430591121
// MI455X (gfx1250) — compile-verified, matrix operand order checked by reading, not run on hardware
//
#include <hip/hip_runtime.h>
#include <hip/hip_bf16.h>

// ---------------------------------------------------------------------------
// Types for CDNA5 WMMA
// ---------------------------------------------------------------------------
typedef __attribute__((ext_vector_type(16))) __bf16 v16bf;
typedef __attribute__((ext_vector_type(8)))  float  v8f;

__device__ __forceinline__ unsigned short f2bf(float f) {
    unsigned int u = __builtin_bit_cast(unsigned int, f);
    unsigned int r = u + 0x7FFFu + ((u >> 16) & 1u);   // round-to-nearest-even
    return (unsigned short)(r >> 16);
}
__device__ __forceinline__ unsigned int f2bf2(float lo, float hi) {
    return (unsigned int)f2bf(lo) | ((unsigned int)f2bf(hi) << 16);
}

// ---------------------------------------------------------------------------
// Generic GEMM: C[M,N] = act( (A[M,K] @ B[K,N]) * scale[n] + bias[n] )
// A,B f32 in global, staged to LDS as bf16 (RNE), f32 accumulate via
// v_wmma_f32_16x16x32_bf16. Block tile 128x128, 8 waves, wave tile 64x32.
// Double-buffered LDS + register-staged software pipeline: global loads for
// tile kb+1 are issued before the WMMAs of tile kb; cvt+ds_store afterwards.
// Fast/slow staging decision is block-uniform (scalar branch, no divergence).
// Requirement: M % 128 == 0 (true for every call site). N,K arbitrary.
// ---------------------------------------------------------------------------
#define GBM 128
#define GBN 128
#define GBK 32
#define GLDA 48   // padded LDS row stride (elements); 96B rows keep 32B align

__global__ __launch_bounds__(256) void gemm_bf16_wmma(
    const float* __restrict__ A, const float* __restrict__ Bm,
    float* __restrict__ C, const float* __restrict__ bias,
    const float* __restrict__ scale, int M, int N, int K, int relu)
{
    __shared__ unsigned short sA[2][GBM * GLDA];
    __shared__ unsigned short sB[2][GBN * GLDA];

    const int tid  = threadIdx.x;
    const int lane = tid & 31;
    const int wave = tid >> 5;
    const int wm   = wave >> 2;     // 0..1 : 64 rows
    const int wn   = wave & 3;      // 0..3 : 32 cols
    const int bm   = blockIdx.y * GBM;
    const int bn   = blockIdx.x * GBN;

    // A staging map: 128 rows x 32 cols, 2 threads/row, 16 f32 each
    const int ar = tid >> 1;
    const int ac = (tid & 1) * 16;
    // B staging map: 32 rows x 128 cols, 8 threads/row, 16 f32 each
    const int br = tid >> 3;
    const int bc = (tid & 7) * 16;

    const bool fullN = (bn + GBN <= N);

    v8f zero = {};
    v8f acc[4][2];
#pragma unroll
    for (int mi = 0; mi < 4; ++mi)
#pragma unroll
        for (int ni = 0; ni < 2; ++ni) acc[mi][ni] = zero;

    float ra[16], rb[16];

    // ---- issue global loads for K-tile at k0 into registers ----
    auto load_regs = [&](int k0) {
        const bool fullK = (k0 + GBK <= K);          // block-uniform
        const float* ap = A + (size_t)(bm + ar) * K + k0 + ac;
        if (fullK) {
            const float4* vp = (const float4*)ap;
            float4 v0 = vp[0], v1 = vp[1], v2 = vp[2], v3 = vp[3];
            ra[0]=v0.x;  ra[1]=v0.y;  ra[2]=v0.z;  ra[3]=v0.w;
            ra[4]=v1.x;  ra[5]=v1.y;  ra[6]=v1.z;  ra[7]=v1.w;
            ra[8]=v2.x;  ra[9]=v2.y;  ra[10]=v2.z; ra[11]=v2.w;
            ra[12]=v3.x; ra[13]=v3.y; ra[14]=v3.z; ra[15]=v3.w;
            if (k0 + 2 * GBK <= K) __builtin_prefetch(ap + GBK, 0, 0);
        } else {
#pragma unroll
            for (int j = 0; j < 16; ++j) {
                int k = k0 + ac + j;
                ra[j] = (k < K) ? ap[j] : 0.0f;
            }
        }
        const int k  = k0 + br;
        const int nb = bn + bc;
        if (fullK && fullN) {
            const float4* bp = (const float4*)(Bm + (size_t)k * N + nb);
            float4 v0 = bp[0], v1 = bp[1], v2 = bp[2], v3 = bp[3];
            rb[0]=v0.x;  rb[1]=v0.y;  rb[2]=v0.z;  rb[3]=v0.w;
            rb[4]=v1.x;  rb[5]=v1.y;  rb[6]=v1.z;  rb[7]=v1.w;
            rb[8]=v2.x;  rb[9]=v2.y;  rb[10]=v2.z; rb[11]=v2.w;
            rb[12]=v3.x; rb[13]=v3.y; rb[14]=v3.z; rb[15]=v3.w;
            if (k + GBK < K) __builtin_prefetch(Bm + (size_t)(k + GBK) * N + nb, 0, 0);
        } else {
#pragma unroll
            for (int j = 0; j < 16; ++j) {
                int n = nb + j;
                rb[j] = (k < K && n < N) ? Bm[(size_t)k * N + n] : 0.0f;
            }
        }
    };

    // ---- convert + store staged registers into LDS buffer ----
    auto store_lds = [&](int buf) {
        unsigned int* da = (unsigned int*)(sA[buf] + ar * GLDA + ac);
#pragma unroll
        for (int j = 0; j < 8; ++j) da[j] = f2bf2(ra[2 * j], ra[2 * j + 1]);
        unsigned short* db = sB[buf];
#pragma unroll
        for (int j = 0; j < 16; ++j) db[(bc + j) * GLDA + br] = f2bf(rb[j]);
    };

    // ---- fragments in contiguous k order (k = 16*(lane/16) + s) + 8 WMMAs ----
    auto compute = [&](int buf) {
        const int lr = lane & 15;
        const int kh = (lane >> 4) * 16;
        v16bf afr[4], bfr[2];
#pragma unroll
        for (int mi = 0; mi < 4; ++mi)
            afr[mi] = *reinterpret_cast<const v16bf*>(
                sA[buf] + (wm * 64 + mi * 16 + lr) * GLDA + kh);
#pragma unroll
        for (int ni = 0; ni < 2; ++ni)
            bfr[ni] = *reinterpret_cast<const v16bf*>(
                sB[buf] + (wn * 32 + ni * 16 + lr) * GLDA + kh);
#pragma unroll
        for (int mi = 0; mi < 4; ++mi)
#pragma unroll
            for (int ni = 0; ni < 2; ++ni)
                acc[mi][ni] = __builtin_amdgcn_wmma_f32_16x16x32_bf16(
                    false, afr[mi], false, bfr[ni], (short)0, acc[mi][ni], false, false);
    };

    const int kTiles = (K + GBK - 1) / GBK;
    load_regs(0);
    store_lds(0);
    __syncthreads();
    for (int kb = 0; kb < kTiles; ++kb) {
        const int cb = kb & 1;
        const bool hasNext = (kb + 1 < kTiles);
        if (hasNext) load_regs((kb + 1) * GBK);   // loads in flight during WMMAs
        compute(cb);
        if (hasNext) store_lds(cb ^ 1);
        __syncthreads();
    }

    // ---- epilogue: C/D layout = lane%16 -> N, lane/16 -> M(+8), vgpr i -> M+i
    const int lr = lane & 15;
    const int lh = (lane >> 4) * 8;
#pragma unroll
    for (int ni = 0; ni < 2; ++ni) {
        int gn = bn + wn * 32 + ni * 16 + lr;
        if (gn >= N) continue;
        float sc = scale ? scale[gn] : 1.0f;
        float bs = bias  ? bias[gn]  : 0.0f;
#pragma unroll
        for (int mi = 0; mi < 4; ++mi) {
            int gm = bm + wm * 64 + mi * 16 + lh;
#pragma unroll
            for (int i = 0; i < 8; ++i) {
                float v = acc[mi][ni][i] * sc + bs;
                if (relu) v = fmaxf(v, 0.0f);
                C[(size_t)(gm + i) * N + gn] = v;
            }
        }
    }
}

// ---------------------------------------------------------------------------
// Elementwise / small kernels (dims hardcoded to this model)
// B=8, T=256, L=2048, E=256, D=512, M=80, VAE=64, K=17, H=8
// ---------------------------------------------------------------------------
__global__ void bn_fuse(const float* cb, const float* bs, const float* bo,
                        float* fsc, float* fbi) {
    int i = blockIdx.x * 256 + threadIdx.x;
    if (i >= 768) return;
    float s = bs[i] * 0.99999500003749969f;   // 1/sqrt(1+1e-5)
    fsc[i] = s;
    fbi[i] = cb[i] * s + bo[i];
}

__global__ void embed_gather(const int* ph, const float* ew, float* out) {
    int idx = blockIdx.x * 256 + threadIdx.x;      // B*T*E = 524288
    int bt = idx >> 8, e = idx & 255;
    out[idx] = ew[ph[bt] * 256 + e];
}

__global__ void im2col_k3(const float* x, float* xc) {
    int idx = blockIdx.x * 256 + threadIdx.x;      // B*T*3E = 1572864
    int bt  = idx / 768;
    int rem = idx - bt * 768;
    int k = rem >> 8;
    int e = rem & 255;
    int b = bt >> 8;
    int t = bt & 255;
    int ts = t + k - 1;
    xc[idx] = (ts >= 0 && ts < 256) ? x[((size_t)(b * 256 + ts)) * 256 + e] : 0.0f;
}

// Bidirectional LSTM, one workgroup per direction; persistent scan over T.
__global__ __launch_bounds__(256) void bilstm(const float* __restrict__ x,
                                              const float* __restrict__ W,
                                              const float* __restrict__ bvec,
                                              const int* __restrict__ len,
                                              float* __restrict__ outp, int dir) {
    __shared__ float shx[8 * 256];
    __shared__ float shh[8 * 256];
    const int tid = threadIdx.x;
    float c[8];
#pragma unroll
    for (int b = 0; b < 8; ++b) { c[b] = 0.f; shh[b * 256 + tid] = 0.f; }

    const float bi = bvec[tid], bg = bvec[256 + tid], bfv = bvec[512 + tid], bo = bvec[768 + tid];

    for (int t = 0; t < 256; ++t) {
        const int ts = dir ? (255 - t) : t;
        for (int i = tid; i < 2048; i += 256) {
            int b = i >> 8, e = i & 255;
            shx[i] = x[((size_t)(b * 256 + ts)) * 256 + e];
        }
        if (dir) {
#pragma unroll
            for (int b = 0; b < 8; ++b)
                if (ts >= len[b] - 1) { c[b] = 0.f; shh[b * 256 + tid] = 0.f; }
        }
        __syncthreads();

        float ai[8] = {0}, ag[8] = {0}, af_[8] = {0}, ao[8] = {0};
        for (int k = 0; k < 256; ++k) {
            float wi = W[k * 1024 + tid];
            float wg = W[k * 1024 + 256 + tid];
            float wf = W[k * 1024 + 512 + tid];
            float wo = W[k * 1024 + 768 + tid];
#pragma unroll
            for (int b = 0; b < 8; ++b) {
                float xv = shx[b * 256 + k];
                ai[b] += wi * xv; ag[b] += wg * xv; af_[b] += wf * xv; ao[b] += wo * xv;
            }
        }
        for (int k = 0; k < 256; ++k) {
            int kk = k + 256;
            float wi = W[kk * 1024 + tid];
            float wg = W[kk * 1024 + 256 + tid];
            float wf = W[kk * 1024 + 512 + tid];
            float wo = W[kk * 1024 + 768 + tid];
#pragma unroll
            for (int b = 0; b < 8; ++b) {
                float hv = shh[b * 256 + k];
                ai[b] += wi * hv; ag[b] += wg * hv; af_[b] += wf * hv; ao[b] += wo * hv;
            }
        }
        __syncthreads();

#pragma unroll
        for (int b = 0; b < 8; ++b) {
            float i_ = 1.f / (1.f + expf(-(ai[b] + bi)));
            float g_ = tanhf(ag[b] + bg);
            float f_ = 1.f / (1.f + expf(-(af_[b] + bfv + 1.0f)));
            float o_ = 1.f / (1.f + expf(-(ao[b] + bo)));
            c[b] = f_ * c[b] + i_ * g_;
            float h_ = o_ * tanhf(c[b]);
            shh[b * 256 + tid] = h_;
            outp[((size_t)(b * 256 + ts)) * 576 + (dir ? 256 : 0) + tid] = h_;
        }
    }
}

__global__ void glu_act(const float* h, float* g) {
    size_t idx = (size_t)blockIdx.x * 256 + threadIdx.x;   // 16384*512
    size_t r = idx >> 9;
    int c = (int)(idx & 511);
    float a = h[r * 1024 + c];
    float b = h[r * 1024 + 512 + c];
    g[idx] = a * (1.0f / (1.0f + expf(-b)));
}

__global__ void lconv_softmax(const float* w, float* ws) {
    int h = threadIdx.x;
    if (h >= 8) return;
    float mx = -1e30f;
    for (int k = 0; k < 17; ++k) mx = fmaxf(mx, w[k * 8 + h]);
    float ek[17], s = 0.f;
    for (int k = 0; k < 17; ++k) { ek[k] = expf(w[k * 8 + h] - mx); s += ek[k]; }
    float inv = 1.0f / s;
    for (int k = 0; k < 17; ++k) ws[k * 8 + h] = ek[k] * inv;
}

__global__ void dwconv17(const float* __restrict__ g, const float* __restrict__ ws,
                         float* __restrict__ y) {
    __shared__ float sw[17 * 8];
    if (threadIdx.x < 136) sw[threadIdx.x] = ws[threadIdx.x];
    __syncthreads();
    size_t idx = (size_t)blockIdx.x * 256 + threadIdx.x;   // 8*2048*512
    int c = (int)(idx & 511);
    size_t bl = idx >> 9;          // b*L + l
    int l = (int)(bl & 2047);
    int h = c >> 6;
    float acc = 0.f;
#pragma unroll
    for (int k = 0; k < 17; ++k) {
        int ll = l + k - 8;
        if (ll >= 0 && ll < 2048)
            acc += sw[k * 8 + h] * g[(bl + (size_t)(k - 8)) * 512 + c];
    }
    y[idx] = acc;
}

// LayerNorm over time axis (axis=1) of (a+b), per (batch, channel); grid = 64
__global__ __launch_bounds__(256) void ln_time_add(const float* __restrict__ a,
                                                   const float* __restrict__ b,
                                                   const float* __restrict__ sc,
                                                   const float* __restrict__ of,
                                                   float* __restrict__ out) {
    __shared__ float s1[256], s2[256], sm[64], sr[64];
    const int bb = blockIdx.x >> 3;
    const int cg = (blockIdx.x & 7) * 64;
    const int cl = threadIdx.x & 63;
    const int lr = threadIdx.x >> 6;
    const int c  = cg + cl;
    float p1 = 0.f, p2 = 0.f;
    for (int l = lr; l < 2048; l += 4) {
        size_t id = ((size_t)bb * 2048 + l) * 512 + c;
        float v = a[id] + b[id];
        p1 += v; p2 += v * v;
    }
    s1[threadIdx.x] = p1; s2[threadIdx.x] = p2;
    __syncthreads();
    if (lr == 0) {
        float t1 = s1[cl] + s1[64 + cl] + s1[128 + cl] + s1[192 + cl];
        float t2 = s2[cl] + s2[64 + cl] + s2[128 + cl] + s2[192 + cl];
        float mean = t1 * (1.0f / 2048.0f);
        float var  = t2 * (1.0f / 2048.0f) - mean * mean;
        sm[cl] = mean;
        sr[cl] = rsqrtf(fmaxf(var, 0.0f) + 1e-5f);
    }
    __syncthreads();
    const float mean = sm[cl], rstd = sr[cl];
    const float scv = sc[c], ofv = of[c];
    for (int l = lr; l < 2048; l += 4) {
        size_t id = ((size_t)bb * 2048 + l) * 512 + c;
        float v = a[id] + b[id];
        out[id] = (v - mean) * rstd * scv + ofv;
    }
}

__global__ void dur_scan(const float* dur, float* stf, float* enf, float* midf) {
    int b = threadIdx.x;
    if (b >= 8) return;
    float e = 0.f;
    for (int t = 0; t < 256; ++t) {
        float d = dur[b * 256 + t];
        e += d;
        stf[b * 256 + t]  = e - d;
        enf[b * 256 + t]  = e;
        midf[b * 256 + t] = e - 0.5f * d;
    }
}

__global__ void seg_pool(const float* __restrict__ m, const float* stf,
                         const float* enf, float* __restrict__ pooled) {
    int r = blockIdx.x;                // b*T + t
    int b = r >> 8;
    float s = stf[r], e = enf[r];
    int lo = max(0, (int)floorf(s));
    int hi = min(2047, (int)ceilf(e));
    int cnt = hi - lo + 1;
    float inv = (cnt > 0) ? 1.0f / (float)cnt : 0.0f;
    for (int d = threadIdx.x; d < 512; d += 256) {
        float acc = 0.f;
        for (int l = lo; l <= hi; ++l)
            acc += m[((size_t)b * 2048 + l) * 512 + d];
        pooled[(size_t)r * 512 + d] = acc * inv;
    }
}

__global__ void vae_reparam(const float* params, const float* noise, float* xcat,
                            float* out_mean, float* out_logstd) {
    int idx = blockIdx.x * 256 + threadIdx.x;   // 2048*64
    int r = idx >> 6, j = idx & 63;
    float mean = params[r * 128 + j];
    float ls   = params[r * 128 + 64 + j];
    out_mean[idx]   = mean;
    out_logstd[idx] = ls;
    xcat[(size_t)r * 576 + 512 + j] = noise[idx] * expf(ls) + mean;
}

__global__ __launch_bounds__(256) void upsample_w(const float* midf, float* Wup) {
    __shared__ float red[256];
    int b = blockIdx.x >> 11;
    int l = blockIdx.x & 2047;
    int t = threadIdx.x;
    float mu = midf[b * 256 + t];
    float df = mu - (float)l;
    float logit = -df * df * 0.1f;
    red[t] = logit; __syncthreads();
    for (int s = 128; s > 0; s >>= 1) { if (t < s) red[t] = fmaxf(red[t], red[t + s]); __syncthreads(); }
    float mx = red[0]; __syncthreads();
    float e = expf(logit - mx);
    red[t] = e; __syncthreads();
    for (int s = 128; s > 0; s >>= 1) { if (t < s) red[t] += red[t + s]; __syncthreads(); }
    float inv = 1.0f / red[0];
    Wup[(size_t)blockIdx.x * 256 + t] = e * inv;
}

// ---------------------------------------------------------------------------
// Launcher
// ---------------------------------------------------------------------------
extern "C" void kernel_launch(void* const* d_in, const int* in_sizes, int n_in,
                              void* d_out, int out_size, void* d_ws, size_t ws_size,
                              hipStream_t stream) {
    (void)in_sizes; (void)n_in; (void)out_size; (void)ws_size;

    const int*   phonemes  = (const int*)  d_in[0];
    const int*   lengths   = (const int*)  d_in[1];
    const float* durations = (const float*)d_in[2];
    const float* mels      = (const float*)d_in[3];
    const float* noise     = (const float*)d_in[4];
    const float* embed_w   = (const float*)d_in[5];
    const float* conv_w    = (const float*)d_in[6];
    const float* conv_b    = (const float*)d_in[7];
    const float* bn_s      = (const float*)d_in[8];
    const float* bn_o      = (const float*)d_in[9];
    const float* lf_w      = (const float*)d_in[10];
    const float* lf_b      = (const float*)d_in[11];
    const float* lb_w      = (const float*)d_in[12];
    const float* lb_b      = (const float*)d_in[13];
    const float* res_in_w  = (const float*)d_in[14];
    const float* res_in_b  = (const float*)d_in[15];
    const float* res_glu_w = (const float*)d_in[16];
    const float* res_glu_b = (const float*)d_in[17];
    const float* res_lconv_w = (const float*)d_in[18];
    const float* res_ln1_s = (const float*)d_in[19];
    const float* res_ln1_o = (const float*)d_in[20];
    const float* res_ff1_w = (const float*)d_in[21];
    const float* res_ff1_b = (const float*)d_in[22];
    const float* res_ff2_w = (const float*)d_in[23];
    const float* res_ff2_b = (const float*)d_in[24];
    const float* res_ln2_s = (const float*)d_in[25];
    const float* res_ln2_o = (const float*)d_in[26];
    const float* vae_w     = (const float*)d_in[27];
    const float* vae_b     = (const float*)d_in[28];
    const float* up_w      = (const float*)d_in[29];
    const float* up_b      = (const float*)d_in[30];
    const float* dec_glu_w = (const float*)d_in[31];
    const float* dec_glu_b = (const float*)d_in[32];
    const float* dec_lconv_w = (const float*)d_in[33];
    const float* dec_ln1_s = (const float*)d_in[34];
    const float* dec_ln1_o = (const float*)d_in[35];
    const float* dec_ff1_w = (const float*)d_in[36];
    const float* dec_ff1_b = (const float*)d_in[37];
    const float* dec_ff2_w = (const float*)d_in[38];
    const float* dec_ff2_b = (const float*)d_in[39];
    const float* dec_ln2_s = (const float*)d_in[40];
    const float* dec_ln2_o = (const float*)d_in[41];
    const float* proj_w    = (const float*)d_in[42];
    const float* proj_b    = (const float*)d_in[43];

    float* out        = (float*)d_out;
    float* out_mean   = out + (size_t)6 * 8 * 2048 * 80;     // 7,864,320
    float* out_logstd = out_mean + (size_t)2048 * 64;

    // ---- workspace bump allocator (floats) ----
    float* w = (float*)d_ws;
    size_t off = 0;
    auto alloc = [&](size_t n) {
        float* p = w + off;
        off += (n + 63) & ~(size_t)63;
        return p;
    };
    float* tokA   = alloc((size_t)2048 * 256);
    float* tokB   = alloc((size_t)2048 * 256);
    float* xcol   = alloc((size_t)2048 * 768);
    float* fsc    = alloc(768);
    float* fbi    = alloc(768);
    float* xcat   = alloc((size_t)2048 * 576);
    float* stf    = alloc(2048);
    float* enf    = alloc(2048);
    float* midf   = alloc(2048);
    float* bufX   = alloc((size_t)16384 * 512);
    float* bufY   = alloc((size_t)16384 * 512);
    float* hglu   = alloc((size_t)16384 * 1024);
    float* gbuf   = alloc((size_t)16384 * 512);
    float* dwbuf  = alloc((size_t)16384 * 512);
    float* ffbuf  = alloc((size_t)4096 * 2048);
    float* pooled = alloc((size_t)2048 * 512);
    float* params = alloc((size_t)2048 * 128);
    float* Wup    = alloc((size_t)16384 * 256);
    float* xup    = alloc((size_t)16384 * 576);
    float* wsmax  = alloc(192);

    auto gemm = [&](const float* Am, const float* Bmat, float* Cm,
                    const float* bias, const float* scale,
                    int M_, int N_, int K_, int relu) {
        dim3 g((N_ + GBN - 1) / GBN, M_ / GBM);
        gemm_bf16_wmma<<<g, 256, 0, stream>>>(Am, Bmat, Cm, bias, scale, M_, N_, K_, relu);
    };

    // ================= Token encoder =================
    bn_fuse<<<3, 256, 0, stream>>>(conv_b, bn_s, bn_o, fsc, fbi);
    embed_gather<<<2048, 256, 0, stream>>>(phonemes, embed_w, tokA);
    const float* cur = tokA;
    float* nxt = tokB;
    for (int i = 0; i < 3; ++i) {
        im2col_k3<<<6144, 256, 0, stream>>>(cur, xcol);
        gemm(xcol, conv_w + (size_t)i * 3 * 256 * 256, nxt,
             fbi + i * 256, fsc + i * 256, 2048, 256, 768, 1);
        const float* tmp = cur; cur = nxt; nxt = (float*)tmp;
    }
    // cur == tokB now; bi-LSTM writes directly into xcat cols [0,512)
    bilstm<<<1, 256, 0, stream>>>(cur, lf_w, lf_b, lengths, xcat, 0);
    bilstm<<<1, 256, 0, stream>>>(cur, lb_w, lb_b, lengths, xcat, 1);

    // ================= LConv block helper =================
    auto lconv_block = [&](const float* glu_w_, const float* glu_b_, const float* lw_,
                           const float* l1s, const float* l1o,
                           const float* f1w, const float* f1b,
                           const float* f2w, const float* f2b,
                           const float* l2s, const float* l2o) {
        lconv_softmax<<<1, 8, 0, stream>>>(lw_, wsmax);
        gemm(bufX, glu_w_, hglu, glu_b_, nullptr, 16384, 1024, 512, 0);
        glu_act<<<32768, 256, 0, stream>>>(hglu, gbuf);
        dwconv17<<<32768, 256, 0, stream>>>(gbuf, wsmax, dwbuf);
        ln_time_add<<<64, 256, 0, stream>>>(dwbuf, bufX, l1s, l1o, bufY);
        for (int c = 0; c < 4; ++c) {   // chunk FF over 4096-row slices
            gemm(bufY + (size_t)c * 4096 * 512, f1w, ffbuf, f1b, nullptr, 4096, 2048, 512, 1);
            gemm(ffbuf, f2w, dwbuf + (size_t)c * 4096 * 512, f2b, nullptr, 4096, 512, 2048, 0);
        }
        ln_time_add<<<64, 256, 0, stream>>>(dwbuf, bufY, l2s, l2o, bufX);
    };

    // ================= Residual encoder =================
    gemm(mels, res_in_w, bufX, res_in_b, nullptr, 16384, 512, 80, 0);
    for (int i = 0; i < 5; ++i)
        lconv_block(res_glu_w + (size_t)i * 512 * 1024, res_glu_b + i * 1024,
                    res_lconv_w + i * 136,
                    res_ln1_s + i * 512, res_ln1_o + i * 512,
                    res_ff1_w + (size_t)i * 512 * 2048, res_ff1_b + i * 2048,
                    res_ff2_w + (size_t)i * 2048 * 512, res_ff2_b + i * 512,
                    res_ln2_s + i * 512, res_ln2_o + i * 512);

    // ================= Duration pooling + VAE =================
    dur_scan<<<1, 8, 0, stream>>>(durations, stf, enf, midf);
    seg_pool<<<2048, 256, 0, stream>>>(bufX, stf, enf, pooled);
    gemm(pooled, vae_w, params, vae_b, nullptr, 2048, 128, 512, 0);
    vae_reparam<<<512, 256, 0, stream>>>(params, noise, xcat, out_mean, out_logstd);

    // ================= Gaussian upsampling =================
    upsample_w<<<16384, 256, 0, stream>>>(midf, Wup);
    for (int b = 0; b < 8; ++b)
        gemm(Wup + (size_t)b * 2048 * 256, xcat + (size_t)b * 256 * 576,
             xup + (size_t)b * 2048 * 576, nullptr, nullptr, 2048, 576, 256, 0);
    gemm(xup, up_w, bufX, up_b, nullptr, 16384, 512, 576, 0);

    // ================= Decoder stack + projections =================
    for (int i = 0; i < 6; ++i) {
        lconv_block(dec_glu_w + (size_t)i * 512 * 1024, dec_glu_b + i * 1024,
                    dec_lconv_w + i * 136,
                    dec_ln1_s + i * 512, dec_ln1_o + i * 512,
                    dec_ff1_w + (size_t)i * 512 * 2048, dec_ff1_b + i * 2048,
                    dec_ff2_w + (size_t)i * 2048 * 512, dec_ff2_b + i * 512,
                    dec_ln2_s + i * 512, dec_ln2_o + i * 512);
        gemm(bufX, proj_w + (size_t)i * 512 * 80, out + (size_t)i * 16384 * 80,
             proj_b + i * 80, nullptr, 16384, 80, 512, 0);
    }
}
